// OutputLayer_16896401342500
// MI455X (gfx1250) — compile-verified, matrix operand order checked by reading, not run on hardware
//
#include <hip/hip_runtime.h>

typedef __attribute__((ext_vector_type(16))) __bf16 v16bf;
typedef __attribute__((ext_vector_type(8)))  float  v8f;
typedef __attribute__((ext_vector_type(4)))  int    v4i;

#define B_ 8
#define S_ 2048
#define D_ 512
#define Y_ 8921
#define YT_PER_BLK 4          // one 16-row y-tile per wave, 4 waves per block
#define NUM_YTILE  560        // ceil(8921/16)=558, padded to 140*4
#define GRID_X     (NUM_YTILE / YT_PER_BLK)   // 140
#define KSTEPS     (D_ / 32)  // 16 WMMA K-steps of 32 (bf16)
#define STILES     (S_ / 16)  // 128 s-tiles
#define XB_BYTES   ((size_t)B_ * S_ * D_ * 2) // bf16 copy of x in workspace

union Frag { uint4 u[2]; v16bf v; };

__device__ inline v16bf frag_zero() {
  v16bf v;
#pragma unroll
  for (int i = 0; i < 16; ++i) v[i] = (__bf16)0.0f;
  return v;
}

// Fragment (16 bf16) from a row-major bf16 row in LDS; p already includes the
// per-lane K offset. Elements 0..7 = K kb..kb+7, elements 8..15 = kb+16..kb+23,
// matching the CDNA5 16-bit A/B matrix VGPR layout.
__device__ inline v16bf frag_from_lds(const __bf16* p) {
  Frag f;
  f.u[0] = *(const uint4*)(p);
  f.u[1] = *(const uint4*)(p + 16);
  return f.v;
}

// Same fragment built from an fp32 row in global memory (convert to bf16).
__device__ inline v16bf frag_from_f32(const float* p) {
  const float4 f0 = *(const float4*)(p);
  const float4 f1 = *(const float4*)(p + 4);
  const float4 f2 = *(const float4*)(p + 16);
  const float4 f3 = *(const float4*)(p + 20);
  v16bf v;
  v[0]  = (__bf16)f0.x; v[1]  = (__bf16)f0.y; v[2]  = (__bf16)f0.z; v[3]  = (__bf16)f0.w;
  v[4]  = (__bf16)f1.x; v[5]  = (__bf16)f1.y; v[6]  = (__bf16)f1.z; v[7]  = (__bf16)f1.w;
  v[8]  = (__bf16)f2.x; v[9]  = (__bf16)f2.y; v[10] = (__bf16)f2.z; v[11] = (__bf16)f2.w;
  v[12] = (__bf16)f3.x; v[13] = (__bf16)f3.y; v[14] = (__bf16)f3.z; v[15] = (__bf16)f3.w;
  return v;
}

// ---------------- async global->LDS staging (CDNA5 GLOBAL_LOAD_ASYNC_TO_LDS) ----
#if __has_builtin(__builtin_amdgcn_global_load_async_to_lds_b128)
#define HAVE_ASYNC_LDS 1
#else
#define HAVE_ASYNC_LDS 0
#endif

__device__ inline void async_fence() {
#if HAVE_ASYNC_LDS
#if __has_builtin(__builtin_amdgcn_s_wait_asynccnt)
  __builtin_amdgcn_s_wait_asynccnt(0);
#else
  asm volatile("s_wait_asynccnt 0x0" ::: "memory");
#endif
#endif
}

// Stage one contiguous 16KB bf16 x-tile into LDS (128 threads x 8 x 16B).
__device__ inline void stage_tile_bf16(const __bf16* __restrict__ src, uint4* dstTile, int tid) {
  const uint4* s = (const uint4*)src;
#if HAVE_ASYNC_LDS
#pragma unroll
  for (int i = 0; i < 8; ++i) {
    __builtin_amdgcn_global_load_async_to_lds_b128(
        (__attribute__((address_space(1))) v4i*)(s + tid + i * 128),
        (__attribute__((address_space(3))) v4i*)(dstTile + tid + i * 128),
        0, 0);
  }
#else
#pragma unroll
  for (int i = 0; i < 8; ++i) dstTile[tid + i * 128] = s[tid + i * 128];
#endif
}

// Stage one 16x512 fp32 x-tile (32KB) into LDS as bf16 (16KB), converting.
__device__ inline void stage_tile_f32(const float* __restrict__ src, uint4* dstTile, int tid) {
  const float4* s = (const float4*)src;
  uint2* d = (uint2*)dstTile;
#pragma unroll
  for (int i = 0; i < 16; ++i) {
    float4 f = s[tid + i * 128];
    union { __bf16 h[4]; uint2 u; } pk;
    pk.h[0] = (__bf16)f.x; pk.h[1] = (__bf16)f.y;
    pk.h[2] = (__bf16)f.z; pk.h[3] = (__bf16)f.w;
    d[tid + i * 128] = pk.u;
  }
}

// ---------------- inner compute: two 16x16 GEMM tiles over K=512 --------------
// Depth-2 software pipeline on the B-fragment LDS loads so DS latency overlaps
// with the WMMA pipe instead of s_wait_dscnt 0 before every matrix op.
__device__ inline void compute_tile(const __bf16* bt, const v16bf* Uf, const v16bf* Wf,
                                    v8f& accA, v8f& accG) {
  v16bf pf0 = frag_from_lds(bt);
  v16bf pf1 = frag_from_lds(bt + 32);
#pragma unroll
  for (int k = 0; k < KSTEPS; ++k) {
    v16bf cur = (k & 1) ? pf1 : pf0;
    if (k + 2 < KSTEPS) {
      v16bf nxt = frag_from_lds(bt + (k + 2) * 32);
      if (k & 1) pf1 = nxt; else pf0 = nxt;
    }
    accA = __builtin_amdgcn_wmma_f32_16x16x32_bf16(false, Uf[k], false, cur,
                                                   (short)0, accA, false, false);
    accG = __builtin_amdgcn_wmma_f32_16x16x32_bf16(false, Wf[k], false, cur,
                                                   (short)0, accG, false, false);
  }
}

// ---------------- online softmax-weighted accumulation over 16 s-columns ------
__device__ inline void softmax_update(const v8f& accA, const v8f& accG,
                                      float* run_max, float* run_sum, float* run_acc) {
#pragma unroll
  for (int r = 0; r < 8; ++r) {
    float a = accA[r], g = accG[r];
    float tmax = a;
#pragma unroll
    for (int m = 8; m >= 1; m >>= 1) tmax = fmaxf(tmax, __shfl_xor(tmax, m, 32));
    const float nm = fmaxf(run_max[r], tmax);
    float p  = __expf(a - nm);
    float pg = p * g;
#pragma unroll
    for (int m = 8; m >= 1; m >>= 1) { p += __shfl_xor(p, m, 32); pg += __shfl_xor(pg, m, 32); }
    const float corr = __expf(run_max[r] - nm);
    run_sum[r] = run_sum[r] * corr + p;
    run_acc[r] = run_acc[r] * corr + pg;
    run_max[r] = nm;
  }
}

// ---------------- x fp32 -> bf16 pre-convert (into workspace) -----------------
__global__ __launch_bounds__(256)
void convert_x_bf16(const float* __restrict__ x, __bf16* __restrict__ xb, int n4) {
  const float4* s = (const float4*)x;
  uint2* d = (uint2*)xb;
  int i = blockIdx.x * blockDim.x + threadIdx.x;
  const int stride = gridDim.x * blockDim.x;
  for (; i < n4; i += stride) {
    float4 f = s[i];
    union { __bf16 h[4]; uint2 u; } pk;
    pk.h[0] = (__bf16)f.x; pk.h[1] = (__bf16)f.y;
    pk.h[2] = (__bf16)f.z; pk.h[3] = (__bf16)f.w;
    d[i] = pk.u;
  }
}

// ---------------- fused per-label attention ----------------------------------
// XBF16 = true: x already bf16 in workspace, staged with async global->LDS DMA.
// XBF16 = false: x fp32 from input, converted during staging (ws fallback).
template <bool XBF16>
__global__ __launch_bounds__(128)
void fused_label_attn(const void* __restrict__ xsrc, const float* __restrict__ Uw,
                      const float* __restrict__ Fw, const float* __restrict__ fb,
                      float* __restrict__ out) {
  __shared__ uint4 tile4[2][1024];   // double-buffered 16x512 bf16 x-tile (2 x 16KB)

  const int tid  = threadIdx.x;
  const int lane = tid & 31;
  const int wave = tid >> 5;
  const int b    = blockIdx.y;
  const int y0   = (blockIdx.x * YT_PER_BLK + wave) * 16;
  const int lrow = lane & 15;                 // M (or N) index within tile
  const int kofs = (lane & 16) ? 8 : 0;       // upper-half lanes hold K+8 chunk

  // ---- resident per-wave weight fragments: U_w and final_w y-tiles ----
  v16bf Uf[KSTEPS], Wf[KSTEPS];
  {
    const int  row   = y0 + lrow;
    const bool valid = row < Y_;
    const float* ur = Uw + (size_t)row * D_;
    const float* wr = Fw + (size_t)row * D_;
#pragma unroll
    for (int k = 0; k < KSTEPS; ++k) {
      const int kb = k * 32 + kofs;
      Uf[k] = valid ? frag_from_f32(ur + kb) : frag_zero();
      Wf[k] = valid ? frag_from_f32(wr + kb) : frag_zero();
    }
  }

  float run_max[8], run_sum[8], run_acc[8];
#pragma unroll
  for (int r = 0; r < 8; ++r) { run_max[r] = -3.0e38f; run_sum[r] = 0.0f; run_acc[r] = 0.0f; }

  const float*  xf = (const float*)xsrc  + (size_t)b * S_ * D_;
  const __bf16* xh = (const __bf16*)xsrc + (size_t)b * S_ * D_;

  if (XBF16) { stage_tile_bf16(xh, tile4[0], tid); async_fence(); }
  else       { stage_tile_f32 (xf, tile4[0], tid); }
  __syncthreads();

  for (int j = 0; j < STILES; ++j) {
    const int cur = j & 1;
    if (j + 1 < STILES) {
      if (XBF16) stage_tile_bf16(xh + (size_t)(j + 1) * 16 * D_, tile4[cur ^ 1], tid);
      else       stage_tile_f32 (xf + (size_t)(j + 1) * 16 * D_, tile4[cur ^ 1], tid);
    }

    const __bf16* bt = (const __bf16*)&tile4[cur][0] + lrow * D_ + kofs;
    v8f accA, accG;
#pragma unroll
    for (int r = 0; r < 8; ++r) { accA[r] = 0.0f; accG[r] = 0.0f; }
    compute_tile(bt, Uf, Wf, accA, accG);
    softmax_update(accA, accG, run_max, run_sum, run_acc);

    if (XBF16) async_fence();
    __syncthreads();
  }

  // ---- write out[b, y] = weighted-avg + bias; lanes 0 and 16 own the rows ----
  if (lrow == 0) {
    const int half = lane >> 4;
#pragma unroll
    for (int r = 0; r < 8; ++r) {
      const int row = y0 + r + 8 * half;
      if (row < Y_) out[(size_t)b * Y_ + row] = run_acc[r] / run_sum[r] + fb[row];
    }
  }
}

extern "C" void kernel_launch(void* const* d_in, const int* in_sizes, int n_in,
                              void* d_out, int out_size, void* d_ws, size_t ws_size,
                              hipStream_t stream) {
  (void)in_sizes; (void)n_in; (void)out_size;
  const float* x  = (const float*)d_in[0];
  const float* Uw = (const float*)d_in[1];
  const float* Fw = (const float*)d_in[2];
  const float* fb = (const float*)d_in[3];
  float* out = (float*)d_out;

  dim3 grid(GRID_X, B_);
  const bool use_bf16_ws = (ws_size >= XB_BYTES) && (d_ws != nullptr);

  if (use_bf16_ws) {
    __bf16* xb = (__bf16*)d_ws;
    const int n4 = (B_ * S_ * D_) / 4;
    convert_x_bf16<<<2048, 256, 0, stream>>>(x, xb, n4);
    fused_label_attn<true><<<grid, 128, 0, stream>>>((const void*)xb, Uw, Fw, fb, out);
  } else {
    fused_label_attn<false><<<grid, 128, 0, stream>>>((const void*)x, Uw, Fw, fb, out);
  }
}
